// EncoderSRNN_81149112091055
// MI455X (gfx1250) — matrix operand order repaired and checked by reading, not compiled, not run on hardware
//
#include <hip/hip_runtime.h>

#define DEVINL __device__ __forceinline__

typedef __attribute__((ext_vector_type(16))) __bf16   v16bf;
typedef __attribute__((ext_vector_type(8)))  float    v8f;
typedef __attribute__((ext_vector_type(4)))  unsigned u32x4;
typedef __attribute__((ext_vector_type(4)))  float    f32x4;
typedef int v4i __attribute__((vector_size(16)));      // matches async-LDS builtin param
typedef __attribute__((address_space(1))) v4i* as1_v4i_p;
typedef __attribute__((address_space(3))) v4i* as3_v4i_p;

// ---- problem dims ----
constexpr int TT = 256, BB = 64, HH = 1024, NS = 2, SS = 64, DD = 4, EE = 128, VV = 10000;
constexpr int K1  = HH + NS * DD * EE;     // 2048 : K of the fused [h | sv] GEMM
constexpr int K2  = K1 / 2;                // 1024 u32 k-pairs per Wcat column
constexpr int H2  = HH / 2;                // 512 u32 k-pairs per 1024-deep column
constexpr int STN = BB * NS * SS * EE;     // 1,048,576 stack elements
constexpr size_t OUT_H  = (size_t)TT * BB * HH;            // offset of h_fin in d_out
constexpr size_t OUT_ST = OUT_H + (size_t)BB * HH;         // offset of st_fin in d_out
constexpr int NWG = 64;                    // persistent grid size
constexpr int NTH = 256;                   // 8 waves per WG (wave32)
constexpr int LDSB = 16 * K1 * 2;          // 64KB A-tile staging per WG

// ---- workspace layout (bytes) ----
constexpr size_t OFF_CNT   = 0;
constexpr size_t OFF_EMB   = 256;                                   // ushort [VV*HH]
constexpr size_t OFF_WI2H  = OFF_EMB   + (size_t)VV * HH * 2;       // u32 [HH][H2]
constexpr size_t OFF_WCAT  = OFF_WI2H  + (size_t)HH * H2 * 4;       // u32 [HH][K2]
constexpr size_t OFF_WACT  = OFF_WCAT  + (size_t)HH * K2 * 4;       // u32 [16][H2]
constexpr size_t OFF_WPUSH = OFF_WACT  + (size_t)16 * H2 * 4;       // u32 [NS*EE][H2]
constexpr size_t OFF_HCAT  = OFF_WPUSH + (size_t)(NS * EE) * H2 * 4;// ushort [2][BB][K1]
constexpr size_t OFF_ST    = OFF_HCAT  + (size_t)2 * BB * K1 * 2;   // f32 [2][STN]
constexpr size_t OFF_ACT   = OFF_ST    + (size_t)2 * STN * 4;       // f32 [BB][16]
constexpr size_t OFF_PUSHV = OFF_ACT   + (size_t)BB * 16 * 4;       // f32 [BB][NS*EE]
constexpr size_t OFF_BIAS  = OFF_PUSHV + (size_t)BB * (NS * EE) * 4;// f32 [HH]

union Frag { u32x4 u[2]; v16bf v; };

DEVINL unsigned short f2bf(float f) {
  unsigned u = __float_as_uint(f);
  return (unsigned short)((u + 0x7FFFu + ((u >> 16) & 1u)) >> 16);   // RNE
}
DEVINL unsigned pk2(float a, float b) {
  return (unsigned)f2bf(a) | ((unsigned)f2bf(b) << 16);
}
DEVINL float sigmoidf(float x) { return 1.0f / (1.0f + __expf(-x)); }

DEVINL v8f wmma_bf16(const Frag& a, const Frag& b, v8f c) {
  // emits v_wmma_f32_16x16x32_bf16
  return __builtin_amdgcn_wmma_f32_16x16x32_bf16(false, a.v, false, b.v,
                                                 (short)0, c, false, false);
}

// A fragment: 16x32 bf16 tile from a row-major bf16 matrix, ISA 16-bit A layout.
// lane half h, row r: VGPR0..3 <- k-pairs (k0+8h)/2 .. +3 ; VGPR4..7 <- (k0+16+8h)/2 ..
DEVINL void load_a(Frag& a, const unsigned short* row, int k0, int half) {
  a.u[0] = *(const u32x4*)(row + k0 + 8 * half);
  a.u[1] = *(const u32x4*)(row + k0 + 16 + 8 * half);
}
// B fragment: 32x16 bf16 tile from column-major packed-pairs u32 matrix [N][K/2].
// lane half h, col c: VGPR0..3 <- k-pairs (k0+8h)/2 .. +3 ; VGPR4..7 <- (k0+16+8h)/2 ..
DEVINL void load_b(Frag& b, const unsigned* col, int k2, int half) {
  b.u[0] = *(const u32x4*)(col + k2 + 4 * half);
  b.u[1] = *(const u32x4*)(col + k2 + 8 + 4 * half);
}

DEVINL void grid_barrier(unsigned* cnt, unsigned target) {
  __syncthreads();
  if (threadIdx.x == 0) {
    __threadfence();
    __hip_atomic_fetch_add(cnt, 1u, __ATOMIC_RELEASE, __HIP_MEMORY_SCOPE_AGENT);
    while (__hip_atomic_load(cnt, __ATOMIC_ACQUIRE, __HIP_MEMORY_SCOPE_AGENT) < target) {
      __builtin_amdgcn_s_sleep(1);
    }
  }
  __syncthreads();
  __threadfence();
}

// ---------------------------------------------------------------------------
// Kernel 1: pack weights to bf16 (B operands as column-major packed k-pairs),
// build initial [h | sv] bf16 operand, copy initial stacks, fold biases,
// reset the grid-barrier counter.
// ---------------------------------------------------------------------------
__global__ void __launch_bounds__(256) pack_kernel(
    const float* emb, const float* W_i2h, const float* b_i2h,
    const float* W_h2h, const float* b_h2h, const float* W_act,
    const float* W_s2h, const float* b_s2h, const float* W_push,
    const float* hidden, const float* stacks0,
    unsigned* cnt, unsigned short* emb_bf, unsigned* Wi2h_p, unsigned* Wcat_p,
    unsigned* Wact_p, unsigned* Wpush_p, unsigned short* hcat0, float* st0,
    float* biassum)
{
  const int tid = blockIdx.x * blockDim.x + threadIdx.x;
  const int nth = gridDim.x * blockDim.x;
  if (tid == 0) *cnt = 0u;

  for (int i = tid; i < VV * HH; i += nth) emb_bf[i] = f2bf(emb[i]);

  for (int i = tid; i < HH * H2; i += nth) {
    int n = i / H2, k = (i % H2) * 2;
    Wi2h_p[i] = pk2(W_i2h[(size_t)k * HH + n], W_i2h[(size_t)(k + 1) * HH + n]);
  }
  // Wcat: rows 0..1023 = W_h2h, rows 1024.. = W_s2h[n][d*E+e][:] (n-major)
  for (int i = tid; i < HH * K2; i += nth) {
    int n = i / K2, k = (i % K2) * 2;
    float a, b;
    if (k < HH) {
      a = W_h2h[(size_t)k * HH + n];
      b = W_h2h[(size_t)(k + 1) * HH + n];
    } else {
      int km = k - HH, sn = km >> 9, rest = km & 511;
      const float* p = W_s2h + (size_t)sn * 512 * HH + (size_t)rest * HH + n;
      a = p[0]; b = p[HH];
    }
    Wcat_p[i] = pk2(a, b);
  }
  // W_act flattened to 16 cols (6 valid: col j -> n=j/3, k=j%3), zero-padded
  for (int i = tid; i < 16 * H2; i += nth) {
    int j = i / H2, k = (i % H2) * 2;
    float a = 0.f, b = 0.f;
    if (j < 6) {
      int n = j / 3, kk = j % 3;
      a = W_act[(size_t)n * HH * 3 + (size_t)k * 3 + kk];
      b = W_act[(size_t)n * HH * 3 + (size_t)(k + 1) * 3 + kk];
    }
    Wact_p[i] = pk2(a, b);
  }
  for (int i = tid; i < (NS * EE) * H2; i += nth) {
    int j = i / H2, k = (i % H2) * 2;
    int n = j >> 7, e = j & 127;
    Wpush_p[i] = pk2(W_push[(size_t)n * HH * EE + (size_t)k * EE + e],
                     W_push[(size_t)n * HH * EE + (size_t)(k + 1) * EE + e]);
  }
  for (int i = tid; i < HH; i += nth)
    biassum[i] = b_i2h[i] + b_h2h[i] + b_s2h[i] + b_s2h[HH + i];

  // initial [h | sv] bf16 operand (buffer 0)
  for (int i = tid; i < BB * K1; i += nth) {
    int b = i / K1, c = i % K1;
    float v;
    if (c < HH) {
      v = hidden[(size_t)b * HH + c];
    } else {
      int cc = c - HH, n = cc >> 9, rest = cc & 511;
      int d = rest >> 7, e = rest & 127;
      v = stacks0[(size_t)b * NS * SS * EE + (size_t)n * SS * EE + (size_t)d * EE + e];
    }
    hcat0[i] = f2bf(v);
  }
  for (int i = tid; i < STN; i += nth) st0[i] = stacks0[i];
}

// ---------------------------------------------------------------------------
// Kernel 2: pre[t,b,:] = emb[inputs[t,b]] @ W_i2h + (b_i2h+b_h2h+sum b_s2h)
// Fused gather + bf16 WMMA GEMM, [16384,1024]x[1024,1024] -> d_out outputs
// region (overwritten in place by h_new during the recurrence).
// ---------------------------------------------------------------------------
__global__ void __launch_bounds__(256) pre_kernel(
    const int* inputs, const unsigned short* emb_bf,
    const unsigned* Wi2h_p, const float* biassum, float* outp)
{
  const int lane = threadIdx.x & 31;
  const int half = lane >> 4;
  const int r    = lane & 15;            // row within M-tile / col within N-tile
  const int gw   = blockIdx.x * 8 + (threadIdx.x >> 5);
  const int nw   = gridDim.x * 8;
  constexpr int NTILES = (TT * BB / 16) * (HH / 16);   // 65536

  for (int tile = gw; tile < NTILES; tile += nw) {
    const int mt = tile >> 6, nt = tile & 63;
    const int frow = mt * 16 + r;                        // flat t*B+b row
    const unsigned short* arow = emb_bf + (size_t)inputs[frow] * HH;
    const unsigned* bcol = Wi2h_p + (size_t)(nt * 16 + r) * H2;
    v8f acc = {};
#pragma unroll 4
    for (int k0 = 0; k0 < HH; k0 += 32) {
      Frag a, b;
      load_a(a, arow, k0, half);
      load_b(b, bcol, k0 >> 1, half);
      acc = wmma_bf16(a, b, acc);
    }
    const float bias = biassum[nt * 16 + r];
#pragma unroll
    for (int i = 0; i < 8; ++i) {
      int row = mt * 16 + i + 8 * half;                  // D layout: M = i + 8*half
      outp[(size_t)row * HH + nt * 16 + r] = acc[i] + bias;
    }
  }
}

// ---------------------------------------------------------------------------
// Kernel 3: persistent recurrent kernel. 64 WGs x 256 threads = 512 waves.
// Per step: stage the WG-shared A tile (16 rows of [h|sv]) into LDS once —
// via GLOBAL_LOAD_ASYNC_TO_LDS_B128 (ASYNCcnt) when the builtin exists,
// else global->VGPR->ds_store. Then phase A = WMMA GEMMs (mid/act/push_val)
// with A from ds_load_b128, phase B = stack shift/blend + next-step bf16
// operand emission. Monotonic grid barrier.
// ---------------------------------------------------------------------------
__global__ void __launch_bounds__(256) srnn_kernel(
    unsigned* cnt,
    unsigned short* hcat,            // [2][BB][K1] bf16  ([h | sv])
    float* stbuf,                    // [2][STN]
    const unsigned* Wcat_p,          // [HH][K2]
    const unsigned* Wact_p,          // [16][H2]
    const unsigned* Wpush_p,         // [NS*EE][H2]
    const float* b_act,              // [NS*3] flat
    const float* b_push,             // [NS*EE] flat
    const float* empty,              // [EE]
    float* actb,                     // [BB][16]
    float* pushv,                    // [BB][NS*EE]
    float* outp)                     // d_out
{
  extern __shared__ unsigned short smem[];               // [16][K1] A-tile stage
  const int lane = threadIdx.x & 31;
  const int half = lane >> 4;
  const int r    = lane & 15;
  const int wid  = blockIdx.x * 8 + (threadIdx.x >> 5);  // 0..511
  const int tid  = blockIdx.x * NTH + threadIdx.x;       // 0..16383
  const int bx   = blockIdx.x;

  // A-tile staging role of this WG (uniform across the block):
  //  bx <  32 : GEMM1 WG, tile row mt = bx>>3, needs full K1 cols
  //  32<=bx<40: GEMM3 WG, tile row mt = (bx-32)>>1, needs HH cols (h part)
  //  bx >= 40 : no staging (GEMM2 waves read global; rest idle in phase A)
  int smt = -1, skc = 0;
  if (bx < 32)      { smt = bx >> 3;        skc = K1; }
  else if (bx < 40) { smt = (bx - 32) >> 1; skc = HH; }

  unsigned bars = 0;

  for (int t = 0; t < TT; ++t) {
    const int cur = t & 1, nxt = cur ^ 1;
    unsigned short* hc = hcat + (size_t)cur * BB * K1;
    unsigned short* hn = hcat + (size_t)nxt * BB * K1;
    const float* sc = stbuf + (size_t)cur * STN;
    float*       sn = stbuf + (size_t)nxt * STN;
    float*       pre = outp + (size_t)t * BB * HH;

    // ---- stage WG-shared A tile into LDS ----
    if (smt >= 0) {
      const int n4row = skc / 8;                          // 16B chunks per row
      for (int i = threadIdx.x; i < 16 * n4row; i += NTH) {
        const int rr = i / n4row, cc = i - rr * n4row;
        const unsigned short* gsrc =
            hc + (size_t)(smt * 16 + rr) * K1 + (size_t)cc * 8;
        unsigned short* ldst = smem + (size_t)i * 8;
#if __has_builtin(__builtin_amdgcn_global_load_async_to_lds_b128)
        // async DMA global -> LDS, no VGPR round-trip (tracked by ASYNCcnt)
        __builtin_amdgcn_global_load_async_to_lds_b128(
            (as1_v4i_p)(v4i*)(void*)gsrc,
            (as3_v4i_p)(v4i*)(void*)ldst, 0, 0);
#else
        *(u32x4*)ldst = *(const u32x4*)gsrc;
#endif
      }
#if __has_builtin(__builtin_amdgcn_global_load_async_to_lds_b128) && \
    __has_builtin(__builtin_amdgcn_s_wait_asynccnt)
      __builtin_amdgcn_s_wait_asynccnt(0);
#endif
    }
    __syncthreads();

    // ------------- phase A: GEMMs off h_t / sv_t -------------
    if (wid < 256) {
      // mid: [64,2048] x [2048,1024], one 16x16 tile per wave, 64 WMMAs
      const int nt = wid & 63;
      const unsigned short* arow = smem + r * K1;         // LDS A operand
      const unsigned* bcol = Wcat_p + (size_t)(nt * 16 + r) * K2;
      v8f acc = {};
#pragma unroll 4
      for (int k0 = 0; k0 < K1; k0 += 32) {
        Frag a, b;
        load_a(a, arow, k0, half);
        load_b(b, bcol, k0 >> 1, half);
        acc = wmma_bf16(a, b, acc);
      }
#pragma unroll
      for (int i = 0; i < 8; ++i) {
        const int bb = smt * 16 + i + 8 * half;
        const size_t o = (size_t)bb * HH + nt * 16 + r;
        const float hv = sigmoidf(acc[i] + pre[o]);
        pre[o] = hv;                                     // outputs[t]
        hn[(size_t)bb * K1 + nt * 16 + r] = f2bf(hv);    // next-step A operand
      }
    } else if (wid < 320) {
      // push_val = sigmoid(h @ W_push + b_push): [64,1024]x[1024,256]
      const int j = wid - 256, nt = j & 15;
      const unsigned short* arow = smem + r * HH;         // LDS A operand (h part)
      const unsigned* bcol = Wpush_p + (size_t)(nt * 16 + r) * H2;
      v8f acc = {};
#pragma unroll 4
      for (int k0 = 0; k0 < HH; k0 += 32) {
        Frag a, b;
        load_a(a, arow, k0, half);
        load_b(b, bcol, k0 >> 1, half);
        acc = wmma_bf16(a, b, acc);
      }
      const float bias = b_push[nt * 16 + r];
#pragma unroll
      for (int i = 0; i < 8; ++i) {
        const int bb = smt * 16 + i + 8 * half;
        pushv[bb * (NS * EE) + nt * 16 + r] = sigmoidf(acc[i] + bias);
      }
    } else if (wid < 324) {
      // act = h @ W_act + b_act (raw linear, 6 valid cols padded to 16)
      const int mt = wid - 320;
      const unsigned short* arow = hc + (size_t)(mt * 16 + r) * K1;  // global A
      const unsigned* bcol = Wact_p + (size_t)r * H2;
      v8f acc = {};
#pragma unroll 4
      for (int k0 = 0; k0 < HH; k0 += 32) {
        Frag a, b;
        load_a(a, arow, k0, half);
        load_b(b, bcol, k0 >> 1, half);
        acc = wmma_bf16(a, b, acc);
      }
      const float bias = (r < 6) ? b_act[r] : 0.0f;
#pragma unroll
      for (int i = 0; i < 8; ++i) {
        const int bb = mt * 16 + i + 8 * half;
        actb[bb * 16 + r] = acc[i] + bias;
      }
    }
    grid_barrier(cnt, (bars += NWG));

    // ------------- phase B: stack shift/blend -------------
    for (int f4 = tid; f4 < STN / 4; f4 += NWG * NTH) {
      const int f = f4 * 4;
      const int e = f & 127;
      const int s = (f >> 7) & 63;
      const int n = (f >> 13) & 1;
      const int bb = f >> 14;
      const size_t base = (size_t)bb * (NS * SS * EE) + (size_t)n * (SS * EE)
                        + (size_t)s * EE + e;
      f32x4 val;
      if (s == SS - 1) {
        val = *(const f32x4*)(empty + e);                // bottom row
      } else {
        const float pp = actb[bb * 16 + n * 3 + 0];
        const float po = actb[bb * 16 + n * 3 + 1];
        const float pn = actb[bb * 16 + n * 3 + 2];
        if (s == 0) {
          const f32x4 pv = *(const f32x4*)(pushv + bb * (NS * EE) + n * EE + e);
          val = pp * pv;
        } else {
          const f32x4 dn = *(const f32x4*)(sc + base - EE);
          const f32x4 up = *(const f32x4*)(sc + base + EE);
          const f32x4 cu = *(const f32x4*)(sc + base);
          val = pp * dn + po * up + pn * cu;
        }
      }
      *(f32x4*)(sn + base) = val;
      if (s < DD) {                                      // sv rows -> next A operand
        unsigned* hp = (unsigned*)(hn + (size_t)bb * K1 + HH + n * 512 + s * 128 + e);
        hp[0] = pk2(val.x, val.y);
        hp[1] = pk2(val.z, val.w);
      }
      if (t == TT - 1) {
        *(f32x4*)(outp + OUT_ST + base) = val;           // st_fin
      }
    }
    grid_barrier(cnt, (bars += NWG));
  }

  // h_fin = outputs[T-1]
  for (int i = tid; i < BB * HH; i += NWG * NTH)
    outp[OUT_H + i] = outp[(size_t)(TT - 1) * BB * HH + i];
}

// ---------------------------------------------------------------------------
extern "C" void kernel_launch(void* const* d_in, const int* in_sizes, int n_in,
                              void* d_out, int out_size, void* d_ws, size_t ws_size,
                              hipStream_t stream) {
  (void)in_sizes; (void)n_in; (void)out_size; (void)ws_size;
  const int*   inputs  = (const int*)d_in[0];
  const float* hidden  = (const float*)d_in[1];
  const float* stacks0 = (const float*)d_in[2];
  // d_in[3] = batch_size (constant)
  const float* emb     = (const float*)d_in[4];
  const float* W_i2h   = (const float*)d_in[5];
  const float* b_i2h   = (const float*)d_in[6];
  const float* W_h2h   = (const float*)d_in[7];
  const float* b_h2h   = (const float*)d_in[8];
  const float* W_act   = (const float*)d_in[9];
  const float* b_act   = (const float*)d_in[10];
  const float* W_s2h   = (const float*)d_in[11];
  const float* b_s2h   = (const float*)d_in[12];
  const float* W_push  = (const float*)d_in[13];
  const float* b_push  = (const float*)d_in[14];
  const float* empty   = (const float*)d_in[15];
  float* outp = (float*)d_out;

  char* ws = (char*)d_ws;
  unsigned*       cnt     = (unsigned*)(ws + OFF_CNT);
  unsigned short* emb_bf  = (unsigned short*)(ws + OFF_EMB);
  unsigned*       Wi2h_p  = (unsigned*)(ws + OFF_WI2H);
  unsigned*       Wcat_p  = (unsigned*)(ws + OFF_WCAT);
  unsigned*       Wact_p  = (unsigned*)(ws + OFF_WACT);
  unsigned*       Wpush_p = (unsigned*)(ws + OFF_WPUSH);
  unsigned short* hcat    = (unsigned short*)(ws + OFF_HCAT);
  float*          stbuf   = (float*)(ws + OFF_ST);
  float*          actb    = (float*)(ws + OFF_ACT);
  float*          pushv   = (float*)(ws + OFF_PUSHV);
  float*          biassum = (float*)(ws + OFF_BIAS);

  pack_kernel<<<512, 256, 0, stream>>>(
      emb, W_i2h, b_i2h, W_h2h, b_h2h, W_act, W_s2h, b_s2h, W_push,
      hidden, stacks0,
      cnt, emb_bf, Wi2h_p, Wcat_p, Wact_p, Wpush_p, hcat, stbuf, biassum);

  pre_kernel<<<1024, 256, 0, stream>>>(inputs, emb_bf, Wi2h_p, biassum, outp);

  srnn_kernel<<<NWG, NTH, LDSB, stream>>>(
      cnt, hcat, stbuf, Wcat_p, Wact_p, Wpush_p,
      b_act, b_push, empty, actb, pushv, outp);
}
